// MultiHeadAttention_17214228922571
// MI455X (gfx1250) — compile-verified, matrix operand order checked by reading, not run on hardware
//
#include <hip/hip_runtime.h>
#include <hip/hip_bf16.h>
#include <math.h>

// MI455X / gfx1250, wave32. Flash attention + projection on v_wmma_f32_16x16x32_f16.
//
// B=2, S=2048, D=1024, H=16, DQ=64. Compute-bound (~43 GFLOP vs ~70MB traffic),
// so everything funnels through WMMA; a cheap memory-bound prep pass (one-time
// fp32->fp16 convert, Q pre-scaled by 1/sqrt(64), V transposed per head) removes
// all per-wave redundant cvt VALU and turns every operand load into b128.
//
// Kernel 2 computes S^T = K*Q^T so softmax output is already in the A-operand
// layout for P*V -- no transpose, only in-lane f16 packing + 2 shfl reductions.

typedef __attribute__((ext_vector_type(16))) _Float16 v16h;
typedef __attribute__((ext_vector_type(8)))  _Float16 v8h;
typedef __attribute__((ext_vector_type(8)))  float    v8f;

#define B_ 2
#define S_ 2048
#define D_ 1024
#define H_ 16
#define DQ_ 64

// f16 WMMA operand load (A or B of v_wmma_*_16x16x32_f16), per-lane pattern:
//   halves[0..7]  = p[ebase+0  .. ebase+7 ]   (one global_load_b128)
//   halves[8..15] = p[ebase+16 .. ebase+23]   (one global_load_b128)
// lanes 0-15 use ebase=0, lanes 16-31 ebase=8 (ISA 16-bit operand layout).
static __device__ inline v16h load_h16(const _Float16* __restrict__ p, int ebase) {
  const v8h lo = *(const v8h*)(p + ebase);
  const v8h hi = *(const v8h*)(p + ebase + 16);
  return __builtin_shufflevector(lo, hi, 0,1,2,3,4,5,6,7,8,9,10,11,12,13,14,15);
}

static __device__ inline v8f wmma16(v16h a, v16h b, v8f c) {
  return __builtin_amdgcn_wmma_f32_16x16x32_f16(false, a, false, b, (short)0, c, false, false);
}

// ---------------------------------------------------------------------------
// Prep: Qh[b,h,s,d] = Q*0.125 (f16), Kh[b,h,s,d] = K (f16),
//       Vt[b,h,d,s] = V (f16, transposed), Woh = Wo (f16, same layout).
// Memory-bound one-shot pass (~80MB total -> ~3us at 23.3 TB/s).
// ---------------------------------------------------------------------------
__global__ void __launch_bounds__(256)
prep_kernel(const float* __restrict__ Q, const float* __restrict__ K,
            const float* __restrict__ V, const float* __restrict__ Wo,
            _Float16* __restrict__ Qh, _Float16* __restrict__ Kh,
            _Float16* __restrict__ Vt, _Float16* __restrict__ Woh) {
  const size_t i = (size_t)blockIdx.x * 256 + threadIdx.x;   // < B*S*D = 4M
  const int d = (int)(i & 63);
  const int h = (int)((i >> 6) & 15);
  const int s = (int)((i >> 10) & 2047);
  const int b = (int)(i >> 21);
  const size_t hq = ((size_t)(b * H_ + h) * S_ + s) * DQ_ + d;
  Qh[hq] = (_Float16)(Q[i] * 0.125f);      // fold 1/sqrt(DQ) into Q
  Kh[hq] = (_Float16)K[i];
  const size_t vt = ((size_t)(b * H_ + h) * DQ_ + d) * S_ + s;
  Vt[vt] = (_Float16)V[i];
  if (i < (size_t)D_ * D_) Woh[i] = (_Float16)Wo[i];
}

// ---------------------------------------------------------------------------
// Flash attention: one wave per 16 queries. ctx written as f16 (proj-ready).
// ---------------------------------------------------------------------------
__global__ void __launch_bounds__(128)
mha_flash_kernel(const _Float16* __restrict__ Qh, const _Float16* __restrict__ Kh,
                 const _Float16* __restrict__ Vt, const int* __restrict__ maskp,
                 _Float16* __restrict__ ctxh) {
  const int lane  = threadIdx.x & 31;
  const int w     = blockIdx.x * 4 + (threadIdx.x >> 5);   // 4096 waves
  const int qtile = w & 127;                               // S/16
  const int h     = (w >> 7) & 15;
  const int b     = w >> 11;
  const int q0    = qtile * 16;
  const bool do_mask = (*maskp) != 0;

  const int l15   = lane & 15;
  const int lhalf = (lane >> 4) & 1;
  const int ebase = lhalf * 8;

  const size_t bh = (size_t)(b * H_ + h);

  // Q as B-operand for S^T = K * Q^T (N = query lane, K = head dim)
  const _Float16* qrow = Qh + (bh * S_ + q0 + l15) * DQ_;
  const v16h bq0 = load_h16(qrow, ebase);
  const v16h bq1 = load_h16(qrow, ebase + 32);

  const _Float16* Kbh = Kh + bh * S_ * DQ_;
  // Vt row of this lane's output column d = c*16 + l15, per d-chunk c
  const _Float16* vt0 = Vt + (bh * DQ_ + (l15))      * (size_t)S_;
  const _Float16* vt1 = Vt + (bh * DQ_ + (16 + l15)) * (size_t)S_;
  const _Float16* vt2 = Vt + (bh * DQ_ + (32 + l15)) * (size_t)S_;
  const _Float16* vt3 = Vt + (bh * DQ_ + (48 + l15)) * (size_t)S_;

  v8f o0 = {}, o1 = {}, o2 = {}, o3 = {};
  float m = -3.0e38f;
  float lsum = 0.0f;
  const int qidx = q0 + l15;

  auto tile = [&](int kt, bool masked) {
    const int key0 = kt * 32;
    // K tile as A-operands (M = key, K = head dim)
    const _Float16* k0p = Kbh + (size_t)(key0 + l15) * DQ_;
    const _Float16* k1p = k0p + 16 * DQ_;
    const v16h ak0_lo = load_h16(k0p, ebase);
    const v16h ak0_hi = load_h16(k0p, ebase + 32);
    const v16h ak1_lo = load_h16(k1p, ebase);
    const v16h ak1_hi = load_h16(k1p, ebase + 32);

    // S^T (pre-scaled logits; Q carried the 1/8)
    v8f st0 = {}, st1 = {};
    st0 = wmma16(ak0_lo, bq0, st0);
    st0 = wmma16(ak0_hi, bq1, st0);
    st1 = wmma16(ak1_lo, bq0, st1);
    st1 = wmma16(ak1_hi, bq1, st1);

    const int kb0 = key0 + lhalf * 8;        // this lane's keys from st0
    const int kb1 = key0 + 16 + lhalf * 8;   // this lane's keys from st1
    float tmax = -3.0e38f;
#pragma unroll
    for (int r = 0; r < 8; ++r) {
      float s0 = st0[r], s1 = st1[r];
      if (masked) {                          // fill(-1e9) before /8  => -1.25e8
        if (kb0 + r >= qidx) s0 = -1.25e8f;
        if (kb1 + r >= qidx) s1 = -1.25e8f;
      }
      st0[r] = s0; st1[r] = s1;
      tmax = fmaxf(tmax, fmaxf(s0, s1));
    }
    tmax = fmaxf(tmax, __shfl_xor(tmax, 16, 32));
    const float mnew = fmaxf(m, tmax);
    const float corr = __expf(m - mnew);

    // probabilities -> A-operand of P*V (in-lane f16 packing only)
    v16h ap;
    float local = 0.0f;
#pragma unroll
    for (int r = 0; r < 8; ++r) {
      const float e0 = __expf(st0[r] - mnew);
      const float e1 = __expf(st1[r] - mnew);
      local += e0 + e1;
      ap[r]     = (_Float16)e0;
      ap[8 + r] = (_Float16)e1;
    }
    lsum = lsum * corr + local + __shfl_xor(local, 16, 32);

    // rescale O (per-query factor broadcast from lane holding that query)
#pragma unroll
    for (int r = 0; r < 8; ++r) {
      const float cr = __shfl(corr, r + lhalf * 8, 32);
      o0[r] *= cr; o1[r] *= cr; o2[r] *= cr; o3[r] *= cr;
    }

    // V tile B-operands: contiguous rows of Vt (keys along the row)
    const v16h bv0 = load_h16(vt0 + key0, ebase);
    const v16h bv1 = load_h16(vt1 + key0, ebase);
    const v16h bv2 = load_h16(vt2 + key0, ebase);
    const v16h bv3 = load_h16(vt3 + key0, ebase);

    o0 = wmma16(ap, bv0, o0);
    o1 = wmma16(ap, bv1, o1);
    o2 = wmma16(ap, bv2, o2);
    o3 = wmma16(ap, bv3, o3);

    m = mnew;
  };

  // Unmasked interior tiles, then masked diagonal tile(s). q0==0 runs the full
  // masked range so the all-masked query 0 matches the reference's uniform
  // softmax over all S keys. Tiles are 32-aligned => no OOB (max key 2047).
  int nfull, nmask_end;
  if (do_mask) {
    if (q0 == 0) { nfull = 0;        nmask_end = S_ / 32; }
    else         { nfull = q0 >> 5;  nmask_end = nfull + 1; }
  } else         { nfull = S_ / 32;  nmask_end = nfull; }

  for (int kt = 0; kt < nfull; ++kt)          tile(kt, false);
  for (int kt = nfull; kt < nmask_end; ++kt)  tile(kt, true);

  // normalize and store ctx as f16 (ready as A-operand for the projection)
#pragma unroll
  for (int r = 0; r < 8; ++r) {
    const float lr   = __shfl(lsum, r + lhalf * 8, 32);
    const float invl = 1.0f / lr;
    _Float16* crow = ctxh + ((size_t)(b * S_ + q0 + r + lhalf * 8) * D_ + h * DQ_ + l15);
    crow[0]  = (_Float16)(o0[r] * invl);
    crow[16] = (_Float16)(o1[r] * invl);
    crow[32] = (_Float16)(o2[r] * invl);
    crow[48] = (_Float16)(o3[r] * invl);
  }
}

// ---------------------------------------------------------------------------
// Projection: Y[i,j] = sum_d ctx[i,d]*Wo[j,d] + bias[j]  (4096x1024x1024 GEMM)
// ---------------------------------------------------------------------------
__global__ void __launch_bounds__(128)
proj_kernel(const _Float16* __restrict__ ctxh, const _Float16* __restrict__ Woh,
            const float* __restrict__ bias, float* __restrict__ Y) {
  const int lane = threadIdx.x & 31;
  const int w    = blockIdx.x * 4 + (threadIdx.x >> 5);   // 16384 waves
  const int ct   = w & 63;                                // D/16 col tiles
  const int rt   = w >> 6;                                // (B*S)/16 row tiles
  const int i0   = rt * 16, j0 = ct * 16;
  const int l15   = lane & 15;
  const int lhalf = (lane >> 4) & 1;
  const int ebase = lhalf * 8;

  const _Float16* arow = ctxh + (size_t)(i0 + l15) * D_;   // A: M = out row
  const _Float16* brow = Woh  + (size_t)(j0 + l15) * D_;   // B: N = out col (Wo^T)

  v8f acc = {};
#pragma unroll 4
  for (int kk = 0; kk < D_; kk += 32) {
    const v16h a = load_h16(arow + kk, ebase);
    const v16h b = load_h16(brow + kk, ebase);
    acc = wmma16(a, b, acc);
  }

  const float bj = bias[j0 + l15];
#pragma unroll
  for (int r = 0; r < 8; ++r) {
    Y[(size_t)(i0 + r + lhalf * 8) * D_ + j0 + l15] = acc[r] + bj;
  }
}

extern "C" void kernel_launch(void* const* d_in, const int* in_sizes, int n_in,
                              void* d_out, int out_size, void* d_ws, size_t ws_size,
                              hipStream_t stream) {
  const float* Q    = (const float*)d_in[0];
  const float* K    = (const float*)d_in[1];
  const float* V    = (const float*)d_in[2];
  const float* Wo_w = (const float*)d_in[3];
  const float* Wo_b = (const float*)d_in[4];
  const int*   mask = (const int*)d_in[5];
  float* Y = (float*)d_out;

  // Workspace layout (f16): Qh, Kh, Vt (4M elems each), Woh (1M), ctxh (4M)
  const size_t NQ = (size_t)B_ * H_ * S_ * DQ_;   // 4,194,304
  _Float16* Qh   = (_Float16*)d_ws;
  _Float16* Kh   = Qh + NQ;
  _Float16* Vt   = Kh + NQ;
  _Float16* Woh  = Vt + NQ;
  _Float16* ctxh = Woh + (size_t)D_ * D_;

  const int prep_blocks = (int)(((size_t)B_ * S_ * D_) / 256);    // 16384
  const int att_waves   = B_ * H_ * (S_ / 16);                    // 4096
  const int proj_waves  = (B_ * S_ / 16) * (D_ / 16);             // 16384

  prep_kernel<<<prep_blocks, 256, 0, stream>>>(Q, K, V, Wo_w, Qh, Kh, Vt, Woh);
  mha_flash_kernel<<<att_waves / 4, 128, 0, stream>>>(Qh, Kh, Vt, mask, ctxh);
  proj_kernel<<<proj_waves / 4, 128, 0, stream>>>(ctxh, Woh, Wo_b, Y);
}
